// BiLSTM_CRF_34067680592090
// MI455X (gfx1250) — matrix operand order repaired and checked by reading, not compiled, not run on hardware
//
#include <hip/hip_runtime.h>
#include <hip/hip_bf16.h>

// ---------------- problem constants ----------------
#define TLEN   8192
#define EMBD   300
#define DPAD   320      // K padded to multiple of 32 for wmma
#define HD     256
#define G4     1024     // 4*HD gates
#define NTAG   19
#define TAG_START 17
#define TAG_STOP  18
#define NEGV  (-10000.0f)

typedef __attribute__((ext_vector_type(16))) __bf16 v16bf;
typedef __attribute__((ext_vector_type(8)))  float  v8f;

struct BPair { uint4 lo; uint4 hi; };   // 32 bytes == v16bf

__device__ __forceinline__ unsigned short f2bf_bits(float f) {
    unsigned int u = __builtin_bit_cast(unsigned int, f);
    unsigned int r = u + 0x7FFFu + ((u >> 16) & 1u);   // round-to-nearest-even
    return (unsigned short)(r >> 16);
}

__device__ __forceinline__ float sigm(float x) { return 1.0f / (1.0f + __expf(-x)); }

// CDNA5 async memory->LDS copy (ASYNCcnt-tracked), per ISA 15.18.3 op 96.
__device__ __forceinline__ void async_ld_b32(unsigned lds_byte_off, const float* gptr) {
    asm volatile("global_load_async_to_lds_b32 %0, %1, off"
                 :: "v"(lds_byte_off), "v"(gptr) : "memory");
}
__device__ __forceinline__ void wait_async0() {
    asm volatile("s_wait_asynccnt 0" ::: "memory");
}

// ---------------------------------------------------------------------------
// Kernel 0a: transpose w_hh [1024,256] -> [256,1024] for coalesced recurrence
// ---------------------------------------------------------------------------
__global__ void whh_transpose_kernel(const float* __restrict__ whh_f,
                                     const float* __restrict__ whh_b,
                                     float* __restrict__ whhT_f,
                                     float* __restrict__ whhT_b) {
    int idx = blockIdx.x * blockDim.x + threadIdx.x;       // 2 * 1024 * 256
    int dir = idx >> 18;
    int r   = idx & 262143;
    int j   = r >> 8;          // gate row  0..1023
    int k   = r & 255;         // hidden col 0..255
    const float* src = dir ? whh_b : whh_f;
    float*       dst = dir ? whhT_b : whhT_f;
    dst[k * G4 + j] = src[j * HD + k];
}

// ---------------------------------------------------------------------------
// Kernel 0b: convert w_ih (both dirs) to bf16, K padded 300 -> 320, once.
// Output layout: wihbf[dir][j][k]  (j = gate row 0..1023, k = 0..319)
// ---------------------------------------------------------------------------
__global__ void wih_bf16_kernel(const float* __restrict__ w_ih_f,
                                const float* __restrict__ w_ih_b,
                                unsigned short* __restrict__ wihbf) {
    int idx = blockIdx.x * blockDim.x + threadIdx.x;       // 2*1024*320
    int dir = idx / (G4 * DPAD);
    int r   = idx % (G4 * DPAD);
    int j   = r / DPAD;
    int k   = r % DPAD;
    const float* w = dir ? w_ih_b : w_ih_f;
    float v = (k < EMBD) ? w[(size_t)j * EMBD + k] : 0.0f;
    wihbf[idx] = f2bf_bits(v);
}

// ---------------------------------------------------------------------------
// Kernel 1: xg[t, j] = emb[sentence[t]] @ w_ih.T + b   (both directions)
// bf16 WMMA, f32 accumulate.  block = 128 threads (4 waves); block tile 16x256
// grid = (T/16, G4/256, 2).  All operand conversion pre-hoisted; inner loop is
// pure vector loads + v_wmma.
// ---------------------------------------------------------------------------
__global__ void xg_gemm_kernel(const int*   __restrict__ sentence,
                               const float* __restrict__ emb,
                               const unsigned short* __restrict__ wihbf,
                               const float* __restrict__ b_f,
                               const float* __restrict__ b_b,
                               float* __restrict__ xg_f,
                               float* __restrict__ xg_b) {
    __shared__ __align__(16) unsigned short lds_a[16 * DPAD]; // bf16 bits, 10 KB

    const int m0    = blockIdx.x * 16;                     // timestep tile
    const int n0blk = blockIdx.y * 256;                    // gate tile
    const int dir   = blockIdx.z;
    const unsigned short* wbf = wihbf + (size_t)dir * G4 * DPAD;
    const float* bias = dir ? b_b  : b_f;
    float*       xg   = dir ? xg_b : xg_f;

    const int tid = threadIdx.x;                           // 0..127

    // cooperative gather of the 16 x 320 activation tile (f32 -> bf16) into LDS
    for (int e = tid; e < 16 * DPAD; e += 128) {
        int m = e / DPAD, k = e % DPAD;
        float v = (k < EMBD) ? emb[(size_t)sentence[m0 + m] * EMBD + k] : 0.0f;
        lds_a[m * DPAD + k] = f2bf_bits(v);
    }
    __syncthreads();

    const int wave  = tid >> 5;
    const int lane  = tid & 31;
    const int laneM = lane & 15;
    const int hiH   = lane >> 4;                           // 0 | 1 lane half

    // 4 x (16x16) accumulators, pre-loaded with bias (same per column)
    v8f acc[4];
    #pragma unroll
    for (int s = 0; s < 4; ++s) {
        float bv = bias[n0blk + wave * 64 + s * 16 + laneM];
        #pragma unroll
        for (int r = 0; r < 8; ++r) acc[s][r] = bv;
    }

    #pragma unroll
    for (int kt = 0; kt < DPAD / 32; ++kt) {
        const int k0 = kt * 32;

        // ---- A fragment (16x32 bf16): two ds_read_b128, ISA lane layout ----
        const unsigned short* ap = &lds_a[laneM * DPAD + k0 + hiH * 8];
        BPair pa;
        pa.lo = *(const uint4*)(ap);        // K chunk +0..7  (this lane half)
        pa.hi = *(const uint4*)(ap + 16);   // K chunk +16..23
        v16bf a = __builtin_bit_cast(v16bf, pa);

        // ---- B fragments (32x16 bf16): two global b128 loads each ----
        #pragma unroll
        for (int s = 0; s < 4; ++s) {
            int jrow = n0blk + wave * 64 + s * 16 + laneM;
            const unsigned short* bp_ = wbf + (size_t)jrow * DPAD + k0 + hiH * 8;
            BPair pb;
            pb.lo = *(const uint4*)(bp_);
            pb.hi = *(const uint4*)(bp_ + 16);
            v16bf b = __builtin_bit_cast(v16bf, pb);
            acc[s] = __builtin_amdgcn_wmma_f32_16x16x32_bf16(
                         false, a, false, b, (short)0, acc[s], false, false);
        }
    }

    // ---- store D: vgpr r -> row (r + 8*hiH), col laneM ----
    #pragma unroll
    for (int s = 0; s < 4; ++s) {
        int ncol = n0blk + wave * 64 + s * 16 + laneM;
        #pragma unroll
        for (int r = 0; r < 8; ++r) {
            int m = r + hiH * 8;
            xg[(size_t)(m0 + m) * G4 + ncol] = acc[s][r];
        }
    }
}

// ---------------------------------------------------------------------------
// Kernel 2: sequential LSTM recurrence, one workgroup per direction.
// 1024 threads: thread j owns gate j; threads 0..255 own c[j]/h[j].
// whhT is [256,1024] so inner-k reads are fully coalesced (L2-resident).
// Next timestep's xg row is double-buffered into LDS with CDNA5 async copies.
// ---------------------------------------------------------------------------
__global__ __launch_bounds__(1024)
void lstm_rec_kernel(const float* __restrict__ xg_f,
                     const float* __restrict__ xg_b,
                     const float* __restrict__ whhT_f,
                     const float* __restrict__ whhT_b,
                     const float* __restrict__ h0,
                     const float* __restrict__ c0,
                     float* __restrict__ hf,
                     float* __restrict__ hb) {
    __shared__ __align__(16) float h_lds[HD];
    __shared__ float g_lds[G4];
    __shared__ float xbuf[2][G4];

    const int dir = blockIdx.x;
    const float* xg   = dir ? xg_b   : xg_f;
    const float* whhT = dir ? whhT_b : whhT_f;
    float*       hout = dir ? hb     : hf;

    const int j = threadIdx.x;                 // 0..1023
    if (j < HD) h_lds[j] = h0[dir * HD + j];
    float c = (j < HD) ? c0[dir * HD + j] : 0.0f;
    __syncthreads();

    // preload step-0 input row into xbuf[0] asynchronously
    {
        const int t0 = dir ? (TLEN - 1) : 0;
        async_ld_b32((unsigned)(size_t)&xbuf[0][j], xg + (size_t)t0 * G4 + j);
    }
    int cur = 0;

    for (int step = 0; step < TLEN; ++step) {
        const int t = dir ? (TLEN - 1 - step) : step;

        // drain our async copy, then immediately kick off next row's copy
        wait_async0();
        const float xval = xbuf[cur][j];       // own element only -> no barrier
        if (step + 1 < TLEN) {
            const int tn = dir ? (t - 1) : (t + 1);
            async_ld_b32((unsigned)(size_t)&xbuf[cur ^ 1][j],
                         xg + (size_t)tn * G4 + j);
        }

        // g[j] = xg[t][j] + sum_k whhT[k][j] * h[k]   (float4 h broadcasts)
        float acc = xval;
        const float* wp = whhT + j;
        #pragma unroll 4
        for (int k = 0; k < HD; k += 4) {
            const float4 hv = *(const float4*)&h_lds[k];
            acc = fmaf(wp[(size_t)(k + 0) * G4], hv.x, acc);
            acc = fmaf(wp[(size_t)(k + 1) * G4], hv.y, acc);
            acc = fmaf(wp[(size_t)(k + 2) * G4], hv.z, acc);
            acc = fmaf(wp[(size_t)(k + 3) * G4], hv.w, acc);
        }
        g_lds[j] = acc;
        __syncthreads();

        if (j < HD) {                           // PyTorch gate order i,f,g,o
            float ig = sigm(g_lds[j]);
            float fg = sigm(g_lds[HD + j]);
            float gg = tanhf(g_lds[2 * HD + j]);
            float og = sigm(g_lds[3 * HD + j]);
            c = fmaf(fg, c, ig * gg);
            float h = og * tanhf(c);
            h_lds[j] = h;
            hout[(size_t)t * HD + j] = h;
        }
        __syncthreads();
        cur ^= 1;
    }
}

// ---------------------------------------------------------------------------
// Kernel 3: feats[t,k] = [hf|hb][t] . w_out[k] + b_out[k]   (f32 VALU; tiny)
// grid = T blocks x 32 threads
// ---------------------------------------------------------------------------
__global__ void feats_kernel(const float* __restrict__ hf,
                             const float* __restrict__ hb,
                             const float* __restrict__ w_out,
                             const float* __restrict__ b_out,
                             float* __restrict__ feats) {
    const int t = blockIdx.x;
    const int k = threadIdx.x;
    if (k >= NTAG) return;
    const float* wrow = w_out + (size_t)k * (2 * HD);
    const float* hfr  = hf + (size_t)t * HD;
    const float* hbr  = hb + (size_t)t * HD;
    float acc = b_out[k];
    #pragma unroll 4
    for (int i = 0; i < HD; ++i) acc = fmaf(hfr[i], wrow[i], acc);
    #pragma unroll 4
    for (int i = 0; i < HD; ++i) acc = fmaf(hbr[i], wrow[HD + i], acc);
    feats[t * NTAG + k] = acc;
}

// ---------------------------------------------------------------------------
// Kernel 4: Viterbi forward scan + batched backtrace. Single wave32.
// out[0] = path score, out[1..T] = best path tags (as float).
// ---------------------------------------------------------------------------
#define BTBATCH 128
__global__ void viterbi_kernel(const float* __restrict__ feats,
                               const float* __restrict__ trans,
                               int* __restrict__ bps,
                               float* __restrict__ out) {
    __shared__ float fv[NTAG];
    __shared__ float fvn[NTAG];
    __shared__ int   bprow[BTBATCH * NTAG];

    const int j = threadIdx.x;                 // 0..31
    if (j < NTAG) fv[j] = (j == TAG_START) ? 0.0f : NEGV;
    __syncthreads();

    for (int t = 0; t < TLEN; ++t) {
        if (j < NTAG) {
            const float* tr = trans + j * NTAG;   // scores[next=j][prev=p]
            float best = fv[0] + tr[0];
            int   bp   = 0;
            #pragma unroll
            for (int p = 1; p < NTAG; ++p) {
                float s = fv[p] + tr[p];
                if (s > best) { best = s; bp = p; }   // first-max tie-break
            }
            fvn[j] = best + feats[t * NTAG + j];
            bps[t * NTAG + j] = bp;
        }
        __syncthreads();
        if (j < NTAG) fv[j] = fvn[j];
        __syncthreads();
    }

    // terminal + score by lane 0
    int tag = 0;
    if (j == 0) {
        float best = fv[0] + trans[TAG_STOP * NTAG + 0];
        int   bt   = 0;
        for (int p = 1; p < NTAG; ++p) {
            float s = fv[p] + trans[TAG_STOP * NTAG + p];
            if (s > best) { best = s; bt = p; }
        }
        out[0] = best;
        tag = bt;
    }

    // batched backtrace: stage 128 bp rows in LDS per round to collapse the
    // global-latency dependent chain
    for (int tb = TLEN - BTBATCH; tb >= 0; tb -= BTBATCH) {
        for (int e = j; e < BTBATCH * NTAG; e += 32)
            bprow[e] = bps[tb * NTAG + e];
        __syncthreads();
        if (j == 0) {
            for (int tt = BTBATCH - 1; tt >= 0; --tt) {
                out[1 + tb + tt] = (float)tag;
                tag = bprow[tt * NTAG + tag];
            }
        }
        __syncthreads();
    }
}

// ---------------------------------------------------------------------------
extern "C" void kernel_launch(void* const* d_in, const int* in_sizes, int n_in,
                              void* d_out, int out_size, void* d_ws, size_t ws_size,
                              hipStream_t stream) {
    (void)in_sizes; (void)n_in; (void)out_size; (void)ws_size;

    const int*   sentence = (const int*)  d_in[0];
    const float* emb      = (const float*)d_in[1];
    const float* w_ih_f   = (const float*)d_in[2];
    const float* w_hh_f   = (const float*)d_in[3];
    const float* b_f      = (const float*)d_in[4];
    const float* w_ih_b   = (const float*)d_in[5];
    const float* w_hh_b   = (const float*)d_in[6];
    const float* b_b      = (const float*)d_in[7];
    const float* w_out    = (const float*)d_in[8];
    const float* b_out    = (const float*)d_in[9];
    const float* trans    = (const float*)d_in[10];
    const float* h0       = (const float*)d_in[11];
    const float* c0       = (const float*)d_in[12];
    float* out = (float*)d_out;

    // workspace carve-up (~86 MB)
    float* xg_f   = (float*)d_ws;
    float* xg_b   = xg_f   + (size_t)TLEN * G4;
    float* whhT_f = xg_b   + (size_t)TLEN * G4;
    float* whhT_b = whhT_f + (size_t)HD * G4;
    float* hf     = whhT_b + (size_t)HD * G4;
    float* hb     = hf     + (size_t)TLEN * HD;
    float* feats  = hb     + (size_t)TLEN * HD;
    int*   bps    = (int*)(feats + (size_t)TLEN * NTAG);
    unsigned short* wihbf = (unsigned short*)(bps + (size_t)TLEN * NTAG);

    // 0a) w_hh transpose (both dirs)
    whh_transpose_kernel<<<(2 * G4 * HD) / 256, 256, 0, stream>>>(
        w_hh_f, w_hh_b, whhT_f, whhT_b);

    // 0b) one-time w_ih f32 -> bf16 (padded) conversion
    wih_bf16_kernel<<<(2 * G4 * DPAD) / 256, 256, 0, stream>>>(
        w_ih_f, w_ih_b, wihbf);

    // 1) bf16 WMMA input GEMMs: xg = emb[sentence] @ w_ih.T + b
    dim3 ggrid(TLEN / 16, G4 / 256, 2);
    xg_gemm_kernel<<<ggrid, 128, 0, stream>>>(
        sentence, emb, wihbf, b_f, b_b, xg_f, xg_b);

    // 2) sequential LSTM recurrence, fwd + bwd concurrently (async LDS feed)
    lstm_rec_kernel<<<2, 1024, 0, stream>>>(
        xg_f, xg_b, whhT_f, whhT_b, h0, c0, hf, hb);

    // 3) tag features
    feats_kernel<<<TLEN, 32, 0, stream>>>(hf, hb, w_out, b_out, feats);

    // 4) Viterbi scan + backtrace
    viterbi_kernel<<<1, 32, 0, stream>>>(feats, trans, bps, out);
}
